// DecoderLayer_30081950941630
// MI455X (gfx1250) — matrix operand order repaired and checked by reading, not compiled, not run on hardware
//
#include <hip/hip_runtime.h>

typedef unsigned short u16;
typedef __attribute__((ext_vector_type(16))) __bf16 v16bf;
typedef __attribute__((ext_vector_type(8)))  float  v8f;

// ---------------------------------------------------------------- helpers ---
__device__ __forceinline__ u16 f2bf(float x) {
  union { float f; unsigned u; } t; t.f = x;
  unsigned r = t.u + 0x7fffu + ((t.u >> 16) & 1u);   // round-to-nearest-even
  return (u16)(r >> 16);
}

// A-matrix fragment (16x32 bf16, MxK): lane holds row m=lane&15.
// VGPR j<4 : k = (lane>>4)*8 + 2j ; VGPR j>=4 : k = 16 + (lane>>4)*8 + 2(j-4)
__device__ __forceinline__ v16bf frag_a(const unsigned* base, int row0, int stride,
                                        int lane, int kpoff) {
  int id = lane & 15, lh = lane >> 4;
  union { v16bf v; unsigned u[8]; } f;
  const unsigned* p = base + (size_t)(row0 + id) * stride + kpoff;
#pragma unroll
  for (int j = 0; j < 8; ++j)
    f.u[j] = p[(j < 4) ? (lh * 4 + j) : (8 + lh * 4 + (j - 4))];
  return f.v;
}

// B-matrix fragment (32x16 bf16, KxN): lane holds col n=lane&15.
// VGPR j<4 : k = lh*8 + 2j ; VGPR j>=4 : k = 16 + lh*8 + 2(j-4)
// LDS/global layout must be [n][k-pairs] (k contiguous per column).
__device__ __forceinline__ v16bf frag_b(const unsigned* base, int col0, int stride,
                                        int lane, int kpoff) {
  int id = lane & 15, lh = lane >> 4;
  union { v16bf v; unsigned u[8]; } f;
  const unsigned* p = base + (size_t)(col0 + id) * stride + kpoff;
#pragma unroll
  for (int j = 0; j < 8; ++j)
    f.u[j] = p[(j < 4) ? (lh * 4 + j) : (8 + lh * 4 + (j - 4))];
  return f.v;
}

__device__ __forceinline__ v8f wmma_bf16(v16bf a, v16bf b, v8f c) {
  return __builtin_amdgcn_wmma_f32_16x16x32_bf16(false, a, false, b, (short)0, c,
                                                 false, false);
}

// ------------------------------------------------------------ tiny kernels ---
__global__ void k_cvt(const float* __restrict__ in, u16* __restrict__ out, int n) {
  int i = blockIdx.x * 256 + threadIdx.x;
  if (i < n) out[i] = f2bf(in[i]);
}

// [H,D,64] fp32 -> bf16 [D, H*64] row-major (K=D, N=512)
__global__ void k_pack_qkv(const float* __restrict__ in, u16* __restrict__ out) {
  int i = blockIdx.x * 256 + threadIdx.x;            // H*D*64 = 262144
  if (i >= 8 * 512 * 64) return;
  int h = i >> 15;
  int d = (i >> 6) & 511;
  int k = i & 63;
  out[d * 512 + h * 64 + k] = f2bf(in[i]);
}

// [R,C] fp32 -> bf16 [C,R]  (transpose pack)
__global__ void k_pack_T(const float* __restrict__ in, u16* __restrict__ out,
                         int R, int C) {
  int i = blockIdx.x * 256 + threadIdx.x;
  if (i >= R * C) return;
  int r = i / C, c = i % C;
  out[(size_t)c * R + r] = f2bf(in[i]);
}

// ------------------------------------------------------------------- GEMM ---
// C[M,N] = A[M,K](bf16,row-major) x B[K,N](bf16,row-major) + bias, epilogues:
// mode 0: row-major out (fp32 and/or bf16), mode 1: bf16 [B,H,S,64] head layout,
// mode 2: bf16 [B,H,64,S] transposed-K layout.  Block: 256 thr, tile 128x64.
__global__ __launch_bounds__(256) void k_gemm(const u16* __restrict__ A,
                                              const u16* __restrict__ Bm,
                                              const float* __restrict__ bias,
                                              float* outF, u16* outB,
                                              int M, int N, int K,
                                              int mode, int relu) {
  __shared__ unsigned sA[128 * 17];    // 128 rows x 16 k-pair dwords (pad 17)
  __shared__ unsigned sB[64 * 17];     // 64 cols  x 16 k-pair dwords (pad 17)
  int t = threadIdx.x, lane = t & 31, wave = t >> 5;
  int wm = wave >> 1, wn = wave & 1;
  int mb = blockIdx.y * 128, nb = blockIdx.x * 64;
  const unsigned* Adw = (const unsigned*)A;
  int Kd = K >> 1;
  v8f acc[2][2] = {};

  for (int kb = 0; kb < K; kb += 32) {
    __syncthreads();
    if (kb + 32 < K)   // hint next A tile into cache (global_prefetch)
      __builtin_prefetch(&Adw[(size_t)(mb + (t >> 1)) * Kd + ((kb + 32) >> 1)], 0, 0);
#pragma unroll
    for (int i = 0; i < 8; ++i) {      // A tile 128x32 bf16 = 2048 dwords
      int e = t + i * 256;
      sA[(e >> 4) * 17 + (e & 15)] =
          Adw[(size_t)(mb + (e >> 4)) * Kd + (kb >> 1) + (e & 15)];
    }
#pragma unroll
    for (int i = 0; i < 4; ++i) {      // B tile 32x64 -> [n][kpair]
      int e = t + i * 256;
      int n = e & 63, kp = e >> 6;
      unsigned lo = Bm[(size_t)(kb + 2 * kp) * N + nb + n];
      unsigned hi = Bm[(size_t)(kb + 2 * kp + 1) * N + nb + n];
      sB[n * 17 + kp] = lo | (hi << 16);
    }
    __syncthreads();
    v16bf a0 = frag_a(sA, wm * 32, 17, lane, 0);
    v16bf a1 = frag_a(sA, wm * 32 + 16, 17, lane, 0);
    v16bf b0 = frag_b(sB, wn * 32, 17, lane, 0);
    v16bf b1 = frag_b(sB, wn * 32 + 16, 17, lane, 0);
    acc[0][0] = wmma_bf16(a0, b0, acc[0][0]);
    acc[0][1] = wmma_bf16(a0, b1, acc[0][1]);
    acc[1][0] = wmma_bf16(a1, b0, acc[1][0]);
    acc[1][1] = wmma_bf16(a1, b1, acc[1][1]);
  }

  int lh = lane >> 4, id = lane & 15;
#pragma unroll
  for (int im = 0; im < 2; ++im)
#pragma unroll
    for (int in_ = 0; in_ < 2; ++in_)
#pragma unroll
      for (int r = 0; r < 8; ++r) {
        int grow = mb + wm * 32 + im * 16 + lh * 8 + r;
        int gcol = nb + wn * 32 + in_ * 16 + id;
        float v = acc[im][in_][r];
        if (bias) v += bias[gcol];
        if (relu) v = fmaxf(v, 0.f);
        if (mode == 0) {
          size_t o = (size_t)grow * N + gcol;
          if (outF) outF[o] = v;
          if (outB) outB[o] = f2bf(v);
        } else {
          int b = grow >> 10, s = grow & 1023;   // S = 1024
          int h = gcol >> 6, dk = gcol & 63;     // head dim 64
          if (mode == 1)
            outB[(((size_t)(b * 8 + h)) * 1024 + s) * 64 + dk] = f2bf(v);
          else
            outB[(((size_t)(b * 8 + h)) * 64 + dk) * 1024 + s] = f2bf(v);
        }
      }
}

// -------------------------------------------------------- flash attention ---
// Q [B,H,S,64] bf16, Kt [B,H,64,S] bf16, V [B,H,S,64] bf16, mask [B,S] int.
// Block = 256 thr (8 waves) handles 128 query rows; streams 32-key tiles.
__global__ __launch_bounds__(256) void k_flash(const u16* __restrict__ Q,
                                               const u16* __restrict__ Kt,
                                               const u16* __restrict__ V,
                                               const int* __restrict__ mask,
                                               u16* __restrict__ out) {
  constexpr int S = 1024;
  __shared__ unsigned sKT[32 * 33];     // [key(32)][d-pairs(32)] pad 33
  __shared__ unsigned sV[64 * 17];      // [v(64)][key-pairs(16)] pad 17
  __shared__ unsigned sP[8 * 16 * 17];  // per-wave P tile [m(16)][key-pairs]
  int t = threadIdx.x, lane = t & 31, wave = t >> 5;
  int id = lane & 15, lh = lane >> 4;
  int qt = blockIdx.x & 7, bh = blockIdx.x >> 3;
  int b = bh >> 3, h = bh & 7;
  int qw = qt * 128 + wave * 16;

  const unsigned* Qdw = (const unsigned*)Q + (size_t)bh * S * 32;
  const u16* Ktp = Kt + (size_t)bh * 64 * S;
  const u16* Vp  = V  + (size_t)bh * S * 64;
  u16* sKT16 = (u16*)sKT;
  u16* sV16  = (u16*)sV;
  unsigned* sPw = sP + wave * 16 * 17;
  u16* sPw16 = (u16*)sPw;

  v16bf qf0 = frag_a(Qdw, qw, 32, lane, 0);    // k = 0..31
  v16bf qf1 = frag_a(Qdw, qw, 32, lane, 16);   // k = 32..63
  v8f Oa[4] = {};
  float Mv[8], Lv[8];
#pragma unroll
  for (int r = 0; r < 8; ++r) { Mv[r] = -3.0e38f; Lv[r] = 0.f; }

  for (int kb = 0; kb < S; kb += 32) {
    __syncthreads();
#pragma unroll
    for (int i = 0; i < 8; ++i) {       // stage K^T (64x32) and V (32x64)
      int e = t + i * 256;
      sKT16[(e & 31) * 66 + (e >> 5)] = Ktp[(size_t)(e >> 5) * S + kb + (e & 31)];
      sV16[(e & 63) * 34 + (e >> 6)] = Vp[(size_t)(kb + (e >> 6)) * 64 + (e & 63)];
    }
    __syncthreads();

    v8f s0 = {}, s1 = {};               // scores: keys kb..+15 / kb+16..+31
    {
      v16bf bk;
      bk = frag_b(sKT, 0, 33, lane, 0);   s0 = wmma_bf16(qf0, bk, s0);
      bk = frag_b(sKT, 16, 33, lane, 0);  s1 = wmma_bf16(qf0, bk, s1);
      bk = frag_b(sKT, 0, 33, lane, 16);  s0 = wmma_bf16(qf1, bk, s0);
      bk = frag_b(sKT, 16, 33, lane, 16); s1 = wmma_bf16(qf1, bk, s1);
    }
    int ok0 = mask[b * S + kb + id];
    int ok1 = mask[b * S + kb + 16 + id];

#pragma unroll
    for (int r = 0; r < 8; ++r) {       // online softmax per query row
      float a0 = ok0 ? s0[r] * 0.125f : -1e10f;
      float a1 = ok1 ? s1[r] * 0.125f : -1e10f;
      float mx = fmaxf(a0, a1);
      mx = fmaxf(mx, __shfl_xor(mx, 1, 32));
      mx = fmaxf(mx, __shfl_xor(mx, 2, 32));
      mx = fmaxf(mx, __shfl_xor(mx, 4, 32));
      mx = fmaxf(mx, __shfl_xor(mx, 8, 32));
      float nm = fmaxf(Mv[r], mx);
      float sc = __expf(Mv[r] - nm);
      Mv[r] = nm;
      float p0 = __expf(a0 - nm);
      float p1 = __expf(a1 - nm);
      float rs = p0 + p1;
      rs += __shfl_xor(rs, 1, 32);
      rs += __shfl_xor(rs, 2, 32);
      rs += __shfl_xor(rs, 4, 32);
      rs += __shfl_xor(rs, 8, 32);
      Lv[r] = Lv[r] * sc + rs;
      Oa[0][r] *= sc; Oa[1][r] *= sc; Oa[2][r] *= sc; Oa[3][r] *= sc;
      int m = r + 8 * lh;               // C-layout row -> P scratch
      sPw16[m * 34 + id] = f2bf(p0);
      sPw16[m * 34 + 16 + id] = f2bf(p1);
    }
    asm volatile("s_wait_dscnt 0x0" ::: "memory");  // wave-local LDS round-trip
    v16bf pf = frag_a(sPw, 0, 17, lane, 0);
#pragma unroll
    for (int vt = 0; vt < 4; ++vt) {
      v16bf vf = frag_b(sV, vt * 16, 17, lane, 0);
      Oa[vt] = wmma_bf16(pf, vf, Oa[vt]);
    }
  }

#pragma unroll
  for (int vt = 0; vt < 4; ++vt)
#pragma unroll
    for (int r = 0; r < 8; ++r) {
      int srow = qw + r + 8 * lh;
      float o = Oa[vt][r] / Lv[r];
      out[(size_t)(b * S + srow) * 512 + h * 64 + vt * 16 + id] = f2bf(o);
    }
}

// ------------------------------------------------------------- layer norm ---
// LN over full [S,D] slice per batch: out = (x0+x1 - mean)*rstd*w + b.
// One block per batch; also emits bf16 copy for the next GEMM A-operand.
__global__ __launch_bounds__(1024) void k_ln(const float* __restrict__ x0,
                                             const float* __restrict__ x1,
                                             const float* __restrict__ w,
                                             const float* __restrict__ bb,
                                             float* __restrict__ outF,
                                             u16* outB) {
  constexpr int PER = 1024 * 512;
  int b = blockIdx.x;
  const float* p0 = x0 + (size_t)b * PER;
  const float* p1 = x1 + (size_t)b * PER;
  float s = 0.f, s2 = 0.f;
  for (int i = threadIdx.x; i < PER; i += 1024) {
    float v = p0[i] + p1[i];
    s += v; s2 += v * v;
  }
  for (int o = 16; o; o >>= 1) { s += __shfl_xor(s, o, 32); s2 += __shfl_xor(s2, o, 32); }
  __shared__ float rs[32], rs2[32];
  __shared__ float smean, srstd;
  int wid = threadIdx.x >> 5, ln = threadIdx.x & 31;
  if (ln == 0) { rs[wid] = s; rs2[wid] = s2; }
  __syncthreads();
  if (threadIdx.x < 32) {
    float a = rs[threadIdx.x], a2 = rs2[threadIdx.x];
    for (int o = 16; o; o >>= 1) { a += __shfl_xor(a, o, 32); a2 += __shfl_xor(a2, o, 32); }
    if (threadIdx.x == 0) {
      float mean = a / (float)PER;
      float var = a2 / (float)PER - mean * mean;
      smean = mean; srstd = rsqrtf(var + 1e-6f);
    }
  }
  __syncthreads();
  float mean = smean, rstd = srstd;
  float* of = outF + (size_t)b * PER;
  u16* ob = outB ? outB + (size_t)b * PER : nullptr;
  for (int i = threadIdx.x; i < PER; i += 1024) {
    float v = (p0[i] + p1[i] - mean) * rstd * w[i] + bb[i];
    of[i] = v;
    if (ob) ob[i] = f2bf(v);
  }
}

// ----------------------------------------------------------------- driver ---
extern "C" void kernel_launch(void* const* d_in, const int* in_sizes, int n_in,
                              void* d_out, int out_size, void* d_ws, size_t ws_size,
                              hipStream_t stream) {
  (void)in_sizes; (void)n_in; (void)out_size; (void)ws_size;
  constexpr int Bn = 8, S = 1024, D = 512, H = 8, DFF = 2048;
  constexpr int M = Bn * S;

  const float* x        = (const float*)d_in[0];
  const float* y        = (const float*)d_in[1];
  const int*   src_mask = (const int*)d_in[2];
  const int*   trg_mask = (const int*)d_in[3];
  const float* m1_wq = (const float*)d_in[4];
  const float* m1_bq = (const float*)d_in[5];
  const float* m1_wk = (const float*)d_in[6];
  const float* m1_bk = (const float*)d_in[7];
  const float* m1_wv = (const float*)d_in[8];
  const float* m1_bv = (const float*)d_in[9];
  const float* m1_wo = (const float*)d_in[10];
  const float* m1_bo = (const float*)d_in[11];
  const float* m2_wq = (const float*)d_in[12];
  const float* m2_bq = (const float*)d_in[13];
  const float* m2_wk = (const float*)d_in[14];
  const float* m2_bk = (const float*)d_in[15];
  const float* m2_wv = (const float*)d_in[16];
  const float* m2_bv = (const float*)d_in[17];
  const float* m2_wo = (const float*)d_in[18];
  const float* m2_bo = (const float*)d_in[19];
  const float* pw1   = (const float*)d_in[20];
  const float* pb1   = (const float*)d_in[21];
  const float* pw2   = (const float*)d_in[22];
  const float* pb2   = (const float*)d_in[23];
  const float* ln1_w = (const float*)d_in[24];
  const float* ln1_b = (const float*)d_in[25];
  const float* ln2_w = (const float*)d_in[26];
  const float* ln2_b = (const float*)d_in[27];
  const float* ln3_w = (const float*)d_in[28];
  const float* ln3_b = (const float*)d_in[29];

  char* wsp = (char*)d_ws;
  auto carve = [&](size_t n) { char* p = wsp; wsp += (n + 255) & ~(size_t)255; return p; };

  u16* xb   = (u16*)carve((size_t)M * D * 2);
  u16* yb   = (u16*)carve((size_t)M * D * 2);
  u16* Wq1  = (u16*)carve((size_t)D * 512 * 2);
  u16* Wk1  = (u16*)carve((size_t)D * 512 * 2);
  u16* Wv1  = (u16*)carve((size_t)D * 512 * 2);
  u16* Wo1  = (u16*)carve((size_t)512 * D * 2);
  u16* Wq2  = (u16*)carve((size_t)D * 512 * 2);
  u16* Wk2  = (u16*)carve((size_t)D * 512 * 2);
  u16* Wv2  = (u16*)carve((size_t)D * 512 * 2);
  u16* Wo2  = (u16*)carve((size_t)512 * D * 2);
  u16* Wp1  = (u16*)carve((size_t)D * DFF * 2);
  u16* Wp2  = (u16*)carve((size_t)DFF * D * 2);
  u16* Qh   = (u16*)carve((size_t)M * 512 * 2);
  u16* Ktb  = (u16*)carve((size_t)M * 512 * 2);
  u16* Vh   = (u16*)carve((size_t)M * 512 * 2);
  u16* attno= (u16*)carve((size_t)M * 512 * 2);
  float* proj = (float*)carve((size_t)M * D * 4);
  float* x1f  = (float*)carve((size_t)M * D * 4);
  u16*   x1b  = (u16*)carve((size_t)M * D * 2);
  float* x2f  = (float*)carve((size_t)M * D * 4);
  u16*   x2b  = (u16*)carve((size_t)M * D * 2);
  u16*   ffh  = (u16*)carve((size_t)M * DFF * 2);
  float* fff  = (float*)carve((size_t)M * D * 4);

  dim3 blk(256);
  int n = M * D;
  k_cvt<<<dim3((n + 255) / 256), blk, 0, stream>>>(x, xb, n);
  k_cvt<<<dim3((n + 255) / 256), blk, 0, stream>>>(y, yb, n);
  int nw = H * D * 64;
  dim3 gpw((nw + 255) / 256);
  k_pack_qkv<<<gpw, blk, 0, stream>>>(m1_wq, Wq1);
  k_pack_qkv<<<gpw, blk, 0, stream>>>(m1_wk, Wk1);
  k_pack_qkv<<<gpw, blk, 0, stream>>>(m1_wv, Wv1);
  k_pack_qkv<<<gpw, blk, 0, stream>>>(m2_wq, Wq2);
  k_pack_qkv<<<gpw, blk, 0, stream>>>(m2_wk, Wk2);
  k_pack_qkv<<<gpw, blk, 0, stream>>>(m2_wv, Wv2);
  k_pack_T<<<dim3((D * 512 + 255) / 256), blk, 0, stream>>>(m1_wo, Wo1, D, 512);
  k_pack_T<<<dim3((D * 512 + 255) / 256), blk, 0, stream>>>(m2_wo, Wo2, D, 512);
  k_pack_T<<<dim3((DFF * D + 255) / 256), blk, 0, stream>>>(pw1, Wp1, DFF, D);
  k_pack_T<<<dim3((D * DFF + 255) / 256), blk, 0, stream>>>(pw2, Wp2, D, DFF);

  dim3 g512(512 / 64, M / 128);
  dim3 g2048(2048 / 64, M / 128);
  dim3 gfa(Bn * H * (S / 128));

  // ---- self attention
  k_gemm<<<g512, blk, 0, stream>>>(xb, Wq1, m1_bq, nullptr, Qh,  M, 512, 512, 1, 0);
  k_gemm<<<g512, blk, 0, stream>>>(xb, Wk1, m1_bk, nullptr, Ktb, M, 512, 512, 2, 0);
  k_gemm<<<g512, blk, 0, stream>>>(xb, Wv1, m1_bv, nullptr, Vh,  M, 512, 512, 1, 0);
  k_flash<<<gfa, blk, 0, stream>>>(Qh, Ktb, Vh, trg_mask, attno);
  k_gemm<<<g512, blk, 0, stream>>>(attno, Wo1, m1_bo, proj, nullptr, M, 512, 512, 0, 0);
  k_ln<<<dim3(Bn), dim3(1024), 0, stream>>>(x, proj, ln1_w, ln1_b, x1f, x1b);

  // ---- cross attention
  k_gemm<<<g512, blk, 0, stream>>>(x1b, Wq2, m2_bq, nullptr, Qh,  M, 512, 512, 1, 0);
  k_gemm<<<g512, blk, 0, stream>>>(yb,  Wk2, m2_bk, nullptr, Ktb, M, 512, 512, 2, 0);
  k_gemm<<<g512, blk, 0, stream>>>(yb,  Wv2, m2_bv, nullptr, Vh,  M, 512, 512, 1, 0);
  k_flash<<<gfa, blk, 0, stream>>>(Qh, Ktb, Vh, src_mask, attno);
  k_gemm<<<g512, blk, 0, stream>>>(attno, Wo2, m2_bo, proj, nullptr, M, 512, 512, 0, 0);
  k_ln<<<dim3(Bn), dim3(1024), 0, stream>>>(x1f, proj, ln2_w, ln2_b, x2f, x2b);

  // ---- FFN
  k_gemm<<<g2048, blk, 0, stream>>>(x2b, Wp1, pb1, nullptr, ffh, M, 2048, 512, 0, 1);
  k_gemm<<<g512,  blk, 0, stream>>>(ffh, Wp2, pb2, fff, nullptr, M, 512, 2048, 0, 0);
  k_ln<<<dim3(Bn), dim3(1024), 0, stream>>>(x2f, fff, ln3_w, ln3_b, (float*)d_out, nullptr);
}
